// KeopsAttentionProduct_16965120819262
// MI455X (gfx1250) — matrix operand order repaired and checked by reading, not compiled, not run on hardware
//
#include <hip/hip_runtime.h>

// ---------------- problem constants (match reference) ----------------
constexpr int Bc = 4, Hc = 16, Tc = 4096, Dc = 64;
constexpr int WIN = 256, STRIDEc = 128, OFFc = 128;
constexpr float NEGV = -1000000000.0f;
constexpr float SCALE = 0.125f;            // 1/sqrt(64)

typedef __attribute__((ext_vector_type(16))) __bf16 v16bf;
typedef __attribute__((ext_vector_type(8)))  float  v8f;

__device__ __forceinline__ v8f wmma_bf16(const v16bf& a, const v16bf& b, const v8f& c) {
    return __builtin_amdgcn_wmma_f32_16x16x32_bf16(
        /*neg_a=*/false, a, /*neg_b=*/false, b,
        /*c_mod=*/(short)0, c, /*reuse_a=*/false, /*reuse_b=*/false);
}

// One workgroup = one (b, h, query-block of 128 rows). 8 waves x 32 lanes.
// Wave w handles query rows [w*16, w*16+16).
__global__ __launch_bounds__(256)
void swin_attn_kernel(const float* __restrict__ Q,
                      const float* __restrict__ K,
                      const float* __restrict__ V,
                      const float* __restrict__ Msk,
                      float* __restrict__ Out)
{
    // LDS: region A = K window as bf16 (256x64) -> later reused as per-wave P staging
    //      region B = V window transposed as bf16 (64 x 256), sVt[d][key]
    __shared__ __align__(16) unsigned char smemA[WIN * Dc * 2];   // 32 KB
    __shared__ __align__(16) unsigned char smemB[Dc * WIN * 2];   // 32 KB

    __bf16* sK  = reinterpret_cast<__bf16*>(smemA);   // sK[key*64 + d]
    __bf16* sVt = reinterpret_cast<__bf16*>(smemB);   // sVt[d*256 + key]

    const int blk  = blockIdx.x;          // 0..31  query block
    const int h    = blockIdx.y;          // head
    const int b    = blockIdx.z;          // batch
    const int tid  = threadIdx.x;
    const int wave = tid >> 5;
    const int lane = tid & 31;
    const int half = lane >> 4;           // 0: lanes 0-15, 1: lanes 16-31
    const int ln   = lane & 15;

    const int start = blk * STRIDEc - OFFc;                 // first key index (may be <0)
    const size_t bh = ((size_t)b * Hc + h) * (size_t)Tc * Dc;

    // ---------------- stage K (row-major bf16) and V (transposed bf16) ----------------
    for (int i = tid; i < (WIN * Dc) / 4; i += 256) {
        int elem = i * 4;
        int key  = elem / Dc;
        int d    = elem % Dc;
        int j    = start + key;
        int jc   = j < 0 ? 0 : (j > Tc - 1 ? Tc - 1 : j);
        float4 kv = *reinterpret_cast<const float4*>(K + bh + (size_t)jc * Dc + d);
        __bf16* dst = sK + key * Dc + d;
        dst[0] = (__bf16)kv.x; dst[1] = (__bf16)kv.y;
        dst[2] = (__bf16)kv.z; dst[3] = (__bf16)kv.w;
    }
    for (int i = tid; i < (WIN * Dc) / 4; i += 256) {
        int elem = i * 4;
        int key  = elem / Dc;
        int d    = elem % Dc;
        int j    = start + key;
        int jc   = j < 0 ? 0 : (j > Tc - 1 ? Tc - 1 : j);
        float4 vv = *reinterpret_cast<const float4*>(V + bh + (size_t)jc * Dc + d);
        sVt[(d + 0) * WIN + key] = (__bf16)vv.x;
        sVt[(d + 1) * WIN + key] = (__bf16)vv.y;
        sVt[(d + 2) * WIN + key] = (__bf16)vv.z;
        sVt[(d + 3) * WIN + key] = (__bf16)vv.w;
    }
    __syncthreads();

    // ---------------- load Q A-fragments (2 K-slices of 32), scale folded in ----------------
    // A 16x32 as loaded here (contiguous k): lane holds row M = lane%16; element m (0..15) maps to
    // K = 16*half + m.
    const int qrow = blk * STRIDEc + wave * 16 + ln;
    const float* qptr = Q + bh + (size_t)qrow * Dc;
    v16bf aQ[2];
#pragma unroll
    for (int ks = 0; ks < 2; ++ks) {
#pragma unroll
        for (int m = 0; m < 16; ++m) {
            int kk = ks * 32 + 16 * half + m;
            aQ[ks][m] = (__bf16)(qptr[kk] * SCALE);
        }
    }

    // ---------------- S = Q * K^T : 16 N-tiles of 16, K-loop of 2 ----------------
    // B 32x16 as loaded here: lane holds column N = lane%16; element m maps to K = 16*half + m.
    v8f S[16];
#pragma unroll
    for (int t = 0; t < 16; ++t) {
        v16bf bK0, bK1;
        const __bf16* kp0 = sK + (t * 16 + ln) * Dc + 0  + 16 * half;
        const __bf16* kp1 = sK + (t * 16 + ln) * Dc + 32 + 16 * half;
#pragma unroll
        for (int m = 0; m < 16; ++m) { bK0[m] = kp0[m]; bK1[m] = kp1[m]; }
        v8f c;
#pragma unroll
        for (int r = 0; r < 8; ++r) c[r] = 0.0f;
        c = wmma_bf16(aQ[0], bK0, c);
        c = wmma_bf16(aQ[1], bK1, c);
        S[t] = c;
    }

    // ---------------- mask + validity (per C-layout lane: key = t*16 + ln) ----------------
#pragma unroll
    for (int t = 0; t < 16; ++t) {
        int j  = start + t * 16 + ln;
        bool valid = (j >= 0) && (j < Tc);
        int jc = j < 0 ? 0 : (j > Tc - 1 ? Tc - 1 : j);
        float mv = Msk[(size_t)b * Tc + jc];
#pragma unroll
        for (int r = 0; r < 8; ++r)
            S[t][r] = valid ? (S[t][r] + mv) : NEGV;
    }

    __syncthreads();   // all waves done reading sK; region A becomes P staging

    // ---------------- softmax over 256 keys (rows live per half-wave) ----------------
    float rmax[8], rsum[8];
#pragma unroll
    for (int r = 0; r < 8; ++r) {
        float m = S[0][r];
#pragma unroll
        for (int t = 1; t < 16; ++t) m = fmaxf(m, S[t][r]);
#pragma unroll
        for (int off = 1; off < 16; off <<= 1)
            m = fmaxf(m, __shfl_xor(m, off, 16));
        rmax[r] = m;
        float s = 0.0f;
#pragma unroll
        for (int t = 0; t < 16; ++t) {
            float p = __expf(S[t][r] - m);
            S[t][r] = p;
            s += p;
        }
#pragma unroll
        for (int off = 1; off < 16; off <<= 1)
            s += __shfl_xor(s, off, 16);
        rsum[r] = s;
    }

    // ---------------- O = P * V : stage P (C-layout -> A-layout) through LDS ----------------
    __bf16* sP = reinterpret_cast<__bf16*>(smemA) + wave * (16 * 32);  // 1 KB per wave

    v8f Oacc[4];
#pragma unroll
    for (int dt = 0; dt < 4; ++dt)
#pragma unroll
        for (int r = 0; r < 8; ++r) Oacc[dt][r] = 0.0f;

#pragma unroll
    for (int kc = 0; kc < 8; ++kc) {          // key chunks of 32
        // write this wave's P chunk (rows 0..15 x cols 0..31) in bf16
#pragma unroll
        for (int tt = 0; tt < 2; ++tt) {
            int t = kc * 2 + tt;
#pragma unroll
            for (int r = 0; r < 8; ++r) {
                int M = r + 8 * half;
                sP[M * 32 + tt * 16 + ln] = (__bf16)S[t][r];
            }
        }
        // read back as A fragment (row M = ln)
        v16bf aP;
#pragma unroll
        for (int m = 0; m < 16; ++m) {
            int kk = 16 * half + m;
            aP[m] = sP[ln * 32 + kk];
        }
        // 4 d-tiles of 16 columns from transposed V
#pragma unroll
        for (int dt = 0; dt < 4; ++dt) {
            v16bf bV;
            const __bf16* vp = sVt + (dt * 16 + ln) * WIN + kc * 32 + 16 * half;
#pragma unroll
            for (int m = 0; m < 16; ++m) bV[m] = vp[m];
            Oacc[dt] = wmma_bf16(aP, bV, Oacc[dt]);
        }
    }

    // ---------------- normalize by row sum and store fp32 ----------------
    const int qbase = blk * STRIDEc + wave * 16;
#pragma unroll
    for (int dt = 0; dt < 4; ++dt) {
#pragma unroll
        for (int r = 0; r < 8; ++r) {
            int M = r + 8 * half;
            float o = Oacc[dt][r] / rsum[r];
            Out[bh + (size_t)(qbase + M) * Dc + dt * 16 + ln] = o;
        }
    }
}

extern "C" void kernel_launch(void* const* d_in, const int* in_sizes, int n_in,
                              void* d_out, int out_size, void* d_ws, size_t ws_size,
                              hipStream_t stream) {
    const float* Q   = (const float*)d_in[0];   // (B,H,T,D) f32
    const float* K   = (const float*)d_in[1];   // (B,H,T,D) f32
    const float* V   = (const float*)d_in[2];   // (B,H,T,D) f32
    const float* Msk = (const float*)d_in[3];   // (B,T) f32
    float* Out       = (float*)d_out;           // (B,H,T,D) f32

    dim3 grid(Tc / STRIDEc, Hc, Bc);            // (32, 16, 4)
    swin_attn_kernel<<<grid, 256, 0, stream>>>(Q, K, V, Msk, Out);
}
